// FrequencyAwareModel_40793599377970
// MI455X (gfx1250) — matrix operand order repaired and checked by reading, not compiled, not run on hardware
//
#include <hip/hip_runtime.h>
#include <hip/hip_bf16.h>

// ---------------- model constants ----------------
#define BB      256
#define SS      100
#define NLOC    200000
#define NUSER   5000
#define DD      128
#define LOCD    48
#define USERD   16
#define TEMPD   64        // D - LOCD - USERD
#define HH      8
#define DH      16
#define FFD     256
#define NLAYERS 3
#define TOPK    2000
#define LN_EPS  1e-5f
#define NTOK    (BB*SS)   // 25600

typedef __attribute__((ext_vector_type(16))) _Float16 v16h;
typedef __attribute__((ext_vector_type(8)))  float    v8f;

// ---------------- helpers ----------------
__device__ __forceinline__ float gelu_exact(float x) {
    return 0.5f * x * (1.0f + erff(x * 0.70710678118654752f));
}

__device__ __forceinline__ float block_reduce_128(float v) {
    // 128 threads = 4 wave32
    __shared__ float sh[4];
    #pragma unroll
    for (int off = 16; off > 0; off >>= 1) v += __shfl_xor(v, off, 32);
    __syncthreads();                       // protect sh across repeated calls
    if ((threadIdx.x & 31) == 0) sh[threadIdx.x >> 5] = v;
    __syncthreads();
    return sh[0] + sh[1] + sh[2] + sh[3];
}

__device__ __forceinline__ void atomic_max_f32(float* addr, float val) {
    unsigned int* ua = (unsigned int*)addr;
    unsigned int old = *ua;
    while (true) {
        float cur = __uint_as_float(old);
        if (cur >= val) break;
        unsigned int assumed = old;
        old = atomicCAS(ua, assumed, __float_as_uint(val));
        if (old == assumed) break;
    }
}

// ---------------- embedding + LN + PE ----------------
// one block (128 threads) per token; thread d owns feature d
__global__ void embed_kernel(const int* __restrict__ loc_seq,
                             const int* __restrict__ user_seq,
                             const int* __restrict__ weekday_seq,
                             const float* __restrict__ start_min,
                             const float* __restrict__ dur,
                             const int* __restrict__ diff_seq,
                             const float* __restrict__ loc_emb,
                             const float* __restrict__ user_emb,
                             const float* __restrict__ Wt,
                             const float* __restrict__ bt,
                             const float* __restrict__ lng,
                             const float* __restrict__ lnb,
                             const float* __restrict__ pe,
                             float* __restrict__ x) {
    int t = blockIdx.x;           // b*S + s
    int s = t % SS;
    int d = threadIdx.x;
    __shared__ float tf[6];
    if (d == 0) {
        const float two_pi = 6.283185307179586f;
        float tr = start_min[t] / 60.0f / 24.0f * two_pi;
        float wd = (float)weekday_seq[t] / 7.0f * two_pi;
        tf[0] = __sinf(tr);
        tf[1] = __cosf(tr);
        tf[2] = log1pf(dur[t]) / 8.0f;
        tf[3] = __sinf(wd);
        tf[4] = __cosf(wd);
        tf[5] = (float)diff_seq[t] / 7.0f;
    }
    __syncthreads();
    float val;
    if (d < LOCD) {
        val = loc_emb[(size_t)loc_seq[t] * LOCD + d];
    } else if (d < LOCD + USERD) {
        val = user_emb[(size_t)user_seq[t] * USERD + (d - LOCD)];
    } else {
        int j = d - LOCD - USERD;
        float acc = bt[j];
        #pragma unroll
        for (int c = 0; c < 6; ++c) acc += tf[c] * Wt[j * 6 + c];
        val = acc;
    }
    float s1 = block_reduce_128(val);
    float s2 = block_reduce_128(val * val);
    float mean = s1 * (1.0f / DD);
    float var  = s2 * (1.0f / DD) - mean * mean;
    float ln = (val - mean) * rsqrtf(var + LN_EPS) * lng[d] + lnb[d];
    x[(size_t)t * DD + d] = ln + pe[s * DD + d];
}

// ---------------- generic WMMA GEMM: C[M,N] = A[M,K] @ W[N,K]^T + bias ----------------
// block = 128 threads = 4 waves; block tile = 32 rows x 64 cols.
// Each wave owns a 32x16 column strip -> 2 WMMAs per K-step sharing one B fragment.
// M must be a multiple of 32; N a multiple of 16; K a multiple of 32.
// act: 0 = none, 1 = exact GELU
__global__ void gemm_wmma_kernel(const float* __restrict__ A,
                                 const float* __restrict__ W,
                                 const float* __restrict__ bias,
                                 float* __restrict__ C,
                                 int M, int N, int K, int act) {
    const int mBase  = blockIdx.y * 32;
    const int wave   = threadIdx.x >> 5;
    const int nTile  = blockIdx.x * 4 + wave;
    const int lane   = threadIdx.x & 31;
    const int laneLo = lane & 15;
    const int laneHi = lane >> 4;
    const bool active = (nTile * 16) < N;
    const int n = nTile * 16 + laneLo;

    __shared__ _Float16 As[32][34];      // 32x32 A tile, padded (row stride 17 dwords)

    v8f acc0 = {};
    v8f acc1 = {};
    for (int k0 = 0; k0 < K; k0 += 32) {
        // prefetch the next K-step's A tile (one 128B row chunk per thread 0..31)
        if (threadIdx.x < 32 && (k0 + 32) < K) {
            __builtin_prefetch(&A[(size_t)(mBase + threadIdx.x) * K + k0 + 32], 0, 1);
        }
        // cooperative stage of A tile (f32 -> f16): 1024 elems / 128 threads
        for (int i = threadIdx.x; i < 32 * 32; i += 128) {
            int r = i >> 5, c = i & 31;
            As[r][c] = (_Float16)A[(size_t)(mBase + r) * K + k0 + c];
        }
        __syncthreads();
        if (active) {
            // B fragment: lane holds column n, K = laneHi*16 + h (contiguous in W row)
            v16h b;
            const float* wrow = W + (size_t)n * K + k0;
            #pragma unroll
            for (int h = 0; h < 16; ++h) b[h] = (_Float16)wrow[(laneHi << 4) + h];
            // A fragments for the two M sub-tiles (K = laneHi*16 + h, matching B)
            v16h a0, a1;
            #pragma unroll
            for (int h = 0; h < 16; ++h) {
                int ka   = (laneHi << 4) + h;
                a0[h] = As[laneLo][ka];
                a1[h] = As[laneLo + 16][ka];
            }
            acc0 = __builtin_amdgcn_wmma_f32_16x16x32_f16(
                false, a0, false, b, (short)0, acc0, false, false);
            acc1 = __builtin_amdgcn_wmma_f32_16x16x32_f16(
                false, a1, false, b, (short)0, acc1, false, false);
        }
        __syncthreads();
    }
    if (active) {
        float bn = bias[n];
        #pragma unroll
        for (int r = 0; r < 8; ++r) {
            // C/D layout: vgpr r, lanes 0-15 -> M=r, lanes 16-31 -> M=r+8
            int m0 = mBase + r + laneHi * 8;
            float v0 = acc0[r] + bn;
            if (act == 1) v0 = gelu_exact(v0);
            C[(size_t)m0 * N + n] = v0;

            int m1 = mBase + 16 + r + laneHi * 8;
            float v1 = acc1[r] + bn;
            if (act == 1) v1 = gelu_exact(v1);
            C[(size_t)m1 * N + n] = v1;
        }
    }
}

// ---------------- attention: one block per (b, h), online two-pass softmax ----------------
__global__ void attn_kernel(const float* __restrict__ qkv,
                            const unsigned char* __restrict__ mask,
                            float* __restrict__ o) {
    int b = blockIdx.x / HH;
    int h = blockIdx.x % HH;
    __shared__ float ks[SS][DH];
    __shared__ float vs[SS][DH];
    for (int i = threadIdx.x; i < SS * DH; i += blockDim.x) {
        int s = i / DH, d = i % DH;
        size_t row = (size_t)(b * SS + s) * (3 * DD);
        ks[s][d] = qkv[row + DD     + h * DH + d];
        vs[s][d] = qkv[row + 2 * DD + h * DH + d];
    }
    __syncthreads();
    int q = threadIdx.x;          // query position
    if (q < SS) {
        float qv[DH];
        size_t qrow = (size_t)(b * SS + q) * (3 * DD) + h * DH;
        #pragma unroll
        for (int d = 0; d < DH; ++d) qv[d] = qkv[qrow + d];

        // pass 1: row max of masked scores
        float mx = -3.0e38f;
        for (int k = 0; k < SS; ++k) {
            float dot = 0.0f;
            #pragma unroll
            for (int d = 0; d < DH; ++d) dot += qv[d] * ks[k][d];
            dot *= 0.25f;                         // 1/sqrt(16)
            if (!mask[b * SS + k]) dot = -1.0e9f;
            mx = fmaxf(mx, dot);
        }
        // pass 2: exp-sum and weighted V accumulation (no scratch array)
        float sum = 0.0f;
        float out[DH];
        #pragma unroll
        for (int d = 0; d < DH; ++d) out[d] = 0.0f;
        for (int k = 0; k < SS; ++k) {
            float dot = 0.0f;
            #pragma unroll
            for (int d = 0; d < DH; ++d) dot += qv[d] * ks[k][d];
            dot *= 0.25f;
            if (!mask[b * SS + k]) dot = -1.0e9f;
            float e = __expf(dot - mx);
            sum += e;
            #pragma unroll
            for (int d = 0; d < DH; ++d) out[d] += e * vs[k][d];
        }
        float inv = 1.0f / sum;
        size_t orow = (size_t)(b * SS + q) * DD + h * DH;
        #pragma unroll
        for (int d = 0; d < DH; ++d) o[orow + d] = out[d] * inv;
    }
}

// ---------------- x = LN(x + y) ----------------
__global__ void add_ln_kernel(float* __restrict__ x,
                              const float* __restrict__ y,
                              const float* __restrict__ g,
                              const float* __restrict__ b) {
    size_t t = blockIdx.x;
    int d = threadIdx.x;
    float val = x[t * DD + d] + y[t * DD + d];
    float s1 = block_reduce_128(val);
    float s2 = block_reduce_128(val * val);
    float mean = s1 * (1.0f / DD);
    float var  = s2 * (1.0f / DD) - mean * mean;
    x[t * DD + d] = (val - mean) * rsqrtf(var + LN_EPS) * g[d] + b[d];
}

// ---------------- gather last valid token per batch ----------------
__global__ void last_kernel(const float* __restrict__ x,
                            const unsigned char* __restrict__ mask,
                            float* __restrict__ last) {
    int b = blockIdx.x;
    __shared__ int sl;
    if (threadIdx.x == 0) {
        int c = 0;
        for (int s = 0; s < SS; ++s) c += mask[b * SS + s] ? 1 : 0;
        sl = c - 1;
    }
    __syncthreads();
    last[(size_t)b * DD + threadIdx.x] = x[(size_t)(b * SS + sl) * DD + threadIdx.x];
}

// ---------------- fill output with -inf ----------------
__global__ void fill_kernel(float* __restrict__ p, size_t n, float v) {
    size_t i = (size_t)blockIdx.x * blockDim.x + threadIdx.x;
    size_t stride = (size_t)gridDim.x * blockDim.x;
    for (; i < n; i += stride) p[i] = v;
}

// ---------------- scatter dense logits to top-k columns ----------------
__global__ void topk_kernel(const float* __restrict__ dlog,
                            const int* __restrict__ idx,
                            float* __restrict__ out) {
    int i = blockIdx.x * blockDim.x + threadIdx.x;    // b*TOPK + j
    if (i >= BB * TOPK) return;
    int b = i / TOPK, j = i % TOPK;
    out[(size_t)b * NLOC + idx[j]] = dlog[i];
}

// ---------------- history scatter-max ----------------
__global__ void hist_kernel(const int* __restrict__ loc_seq,
                            const unsigned char* __restrict__ mask,
                            const float* __restrict__ loc_emb,
                            const float* __restrict__ query,
                            float* __restrict__ out) {
    int i = blockIdx.x * blockDim.x + threadIdx.x;    // b*S + s
    if (i >= NTOK) return;
    if (!mask[i]) return;
    int b = i / SS;
    int loc = loc_seq[i];
    const float* le = loc_emb + (size_t)loc * LOCD;
    const float* q  = query + (size_t)b * LOCD;
    float sc = 0.0f;
    #pragma unroll
    for (int d = 0; d < LOCD; ++d) sc += le[d] * q[d];
    atomic_max_f32(out + (size_t)b * NLOC + loc, sc);
}

// ---------------- launcher ----------------
extern "C" void kernel_launch(void* const* d_in, const int* in_sizes, int n_in,
                              void* d_out, int out_size, void* d_ws, size_t ws_size,
                              hipStream_t stream) {
    const int*   loc_seq   = (const int*)d_in[0];
    const int*   user_seq  = (const int*)d_in[1];
    const int*   weekday   = (const int*)d_in[2];
    const float* start_min = (const float*)d_in[3];
    const float* dur       = (const float*)d_in[4];
    const int*   diff_seq  = (const int*)d_in[5];
    const unsigned char* mask = (const unsigned char*)d_in[6];
    const float* loc_emb   = (const float*)d_in[7];
    const float* user_emb  = (const float*)d_in[8];
    const float* Wt        = (const float*)d_in[9];
    const float* bt        = (const float*)d_in[10];
    const float* lng       = (const float*)d_in[11];
    const float* lnb       = (const float*)d_in[12];
    const float* pe        = (const float*)d_in[13];
    const float* Wqkv      = (const float*)d_in[14];
    const float* bqkv      = (const float*)d_in[15];
    const float* Wo        = (const float*)d_in[16];
    const float* bo        = (const float*)d_in[17];
    const float* ln1g      = (const float*)d_in[18];
    const float* ln1b      = (const float*)d_in[19];
    const float* W1        = (const float*)d_in[20];
    const float* b1        = (const float*)d_in[21];
    const float* W2        = (const float*)d_in[22];
    const float* b2        = (const float*)d_in[23];
    const float* ln2g      = (const float*)d_in[24];
    const float* ln2b      = (const float*)d_in[25];
    const float* Wd1       = (const float*)d_in[26];
    const float* bd1       = (const float*)d_in[27];
    const float* Wd2       = (const float*)d_in[28];
    const float* bd2       = (const float*)d_in[29];
    const float* Wc        = (const float*)d_in[30];
    const float* bc        = (const float*)d_in[31];
    const int*   topk_idx  = (const int*)d_in[32];

    float* out = (float*)d_out;

    // workspace carve-out (floats)
    float* ws = (float*)d_ws;
    float* x     = ws;                       ws += (size_t)NTOK * DD;       // 3.28M
    float* qkv   = ws;                       ws += (size_t)NTOK * 3 * DD;   // 9.83M (also reused as tmp)
    float* obuf  = ws;                       ws += (size_t)NTOK * DD;       // 3.28M
    float* hbuf  = ws;                       ws += (size_t)NTOK * FFD;      // 6.55M
    float* lastb = ws;                       ws += (size_t)BB * DD;
    float* t1    = ws;                       ws += (size_t)BB * FFD;
    float* dlog  = ws;                       ws += (size_t)BB * TOPK;
    float* query = ws;                       ws += (size_t)BB * LOCD;
    (void)ws_size; (void)n_in; (void)in_sizes; (void)out_size;

    // 1) embedding + LN + positional encoding
    embed_kernel<<<NTOK, 128, 0, stream>>>(loc_seq, user_seq, weekday, start_min,
                                           dur, diff_seq, loc_emb, user_emb,
                                           Wt, bt, lng, lnb, pe, x);

    const int mBlk = NTOK / 32;     // 800 row-blocks of 32
    // 2) transformer layers
    for (int l = 0; l < NLAYERS; ++l) {
        const float* Wqkv_l = Wqkv + (size_t)l * 3 * DD * DD;
        const float* bqkv_l = bqkv + (size_t)l * 3 * DD;
        const float* Wo_l   = Wo   + (size_t)l * DD * DD;
        const float* bo_l   = bo   + (size_t)l * DD;
        const float* W1_l   = W1   + (size_t)l * FFD * DD;
        const float* b1_l   = b1   + (size_t)l * FFD;
        const float* W2_l   = W2   + (size_t)l * DD * FFD;
        const float* b2_l   = b2   + (size_t)l * DD;

        gemm_wmma_kernel<<<dim3(6, mBlk), 128, 0, stream>>>(
            x, Wqkv_l, bqkv_l, qkv, NTOK, 3 * DD, DD, 0);

        attn_kernel<<<BB * HH, 128, 0, stream>>>(qkv, mask, obuf);

        // Wo projection (qkv buffer reused as tmp output)
        gemm_wmma_kernel<<<dim3(2, mBlk), 128, 0, stream>>>(
            obuf, Wo_l, bo_l, qkv, NTOK, DD, DD, 0);
        add_ln_kernel<<<NTOK, 128, 0, stream>>>(x, qkv, ln1g + l * DD, ln1b + l * DD);

        gemm_wmma_kernel<<<dim3(4, mBlk), 128, 0, stream>>>(
            x, W1_l, b1_l, hbuf, NTOK, FFD, DD, 1);
        gemm_wmma_kernel<<<dim3(2, mBlk), 128, 0, stream>>>(
            hbuf, W2_l, b2_l, qkv, NTOK, DD, FFD, 0);
        add_ln_kernel<<<NTOK, 128, 0, stream>>>(x, qkv, ln2g + l * DD, ln2b + l * DD);
    }

    // 3) head
    last_kernel<<<BB, 128, 0, stream>>>(x, mask, lastb);
    gemm_wmma_kernel<<<dim3(4, BB / 32), 128, 0, stream>>>(
        lastb, Wd1, bd1, t1, BB, FFD, DD, 1);
    gemm_wmma_kernel<<<dim3(32, BB / 32), 128, 0, stream>>>(
        t1, Wd2, bd2, dlog, BB, TOPK, FFD, 0);
    gemm_wmma_kernel<<<dim3(1, BB / 32), 128, 0, stream>>>(
        lastb, Wc, bc, query, BB, LOCD, DD, 0);

    // 4) final output: fill -inf, scatter top-k logits, then history atomic-max
    size_t total = (size_t)BB * NLOC;
    fill_kernel<<<4096, 256, 0, stream>>>(out, total, -__builtin_inff());
    topk_kernel<<<(BB * TOPK + 255) / 256, 256, 0, stream>>>(dlog, topk_idx, out);
    hist_kernel<<<(NTOK + 255) / 256, 256, 0, stream>>>(loc_seq, mask, loc_emb, query, out);
}
